// GPT_8134668059158
// MI455X (gfx1250) — compile-verified, matrix operand order checked by reading, not run on hardware
//
#include <hip/hip_runtime.h>
#include <hip/hip_bf16.h>
#include <math.h>

// ---------------------------------------------------------------------------
// GPT forward (single block) for gfx1250 / MI455X.
// All GEMMs: v_wmma_f32_16x16x32_bf16, LDS double-buffered 128x128 block tiles
// filled with GLOBAL_LOAD_ASYNC_TO_LDS_B128 when the toolchain exposes it.
// ---------------------------------------------------------------------------

typedef __bf16 bf16_t;
typedef __attribute__((ext_vector_type(16))) __bf16 v16bf;
typedef __attribute__((ext_vector_type(8)))  __bf16 v8bf;
typedef __attribute__((ext_vector_type(8)))  float  v8f;

#define GPT_V  50257
#define GPT_VP 50304   /* V padded to multiple of 128 (block N tile) */
#define GPT_D  768
#define GPT_H  3072
#define GPT_S  1024
#define GPT_B  4
#define GPT_N  4096    /* B*S tokens */
#define LN_EPS 1e-5f

#define BT   128       /* block tile (M and N) */
#define KT   32        /* K slab per WMMA step */
#define TPAD 40        /* LDS row stride in bf16 elems (80B: 16B-aligned, bank-skewed) */

// ---------------- async global->LDS copy (CDNA5), with sync fallback -------
#if __has_builtin(__builtin_amdgcn_global_load_async_to_lds_b128)
#define GPT_ASYNC_LDS 1
typedef int gpt4i __attribute__((vector_size(16)));
typedef __attribute__((address_space(1))) gpt4i* gas1_p;   // global int4*
typedef __attribute__((address_space(3))) gpt4i* las3_p;   // LDS int4*
#else
#define GPT_ASYNC_LDS 0
#endif

__device__ __forceinline__ void cp16_g2l(bf16_t* lds, const bf16_t* g) {
#if GPT_ASYNC_LDS
    __builtin_amdgcn_global_load_async_to_lds_b128((gas1_p)(void*)g,
                                                   (las3_p)(void*)lds, 0, 0);
#else
    *reinterpret_cast<v8bf*>(lds) = *reinterpret_cast<const v8bf*>(g);
#endif
}

__device__ __forceinline__ void wait_async_fills() {
#if GPT_ASYNC_LDS
#if __has_builtin(__builtin_amdgcn_s_wait_asynccnt)
    __builtin_amdgcn_s_wait_asynccnt(0);
#else
    asm volatile("s_wait_asynccnt 0x0" ::: "memory");
#endif
#endif
}

// Stage a BT x KT tile: dst rows of TPAD elems; 512 16B chunks over 256 thr.
__device__ __forceinline__ void fill_tile(bf16_t* dst, const bf16_t* src,
                                          int row0, int ld, int kk, int tid) {
#pragma unroll
    for (int c = tid; c < BT * 4; c += 256) {
        const int row = c >> 2;
        const int p   = (c & 3) * 8;
        cp16_g2l(dst + row * TPAD + p,
                 src + (size_t)(row0 + row) * (size_t)ld + kk + p);
    }
}

// ---------------------------------------------------------------------------
// 16-bit WMMA operand fragment from an LDS tile (ISA 7.12.2 layout):
// lane l (0..15 | 16..31) -> row rrel+(l&15), K base (l>>4)*8;
// elems 0..7 = k kb..kb+7, elems 8..15 = k kb+16..kb+23 (two b128 DS reads).
// ---------------------------------------------------------------------------
__device__ __forceinline__ v16bf lds_frag(const bf16_t* buf, int rrel, int lane) {
    const int r  = rrel + (lane & 15);
    const int kb = (lane >> 4) << 3;
    const bf16_t* p = buf + r * TPAD + kb;
    v8bf lo = *reinterpret_cast<const v8bf*>(p);
    v8bf hi = *reinterpret_cast<const v8bf*>(p + 16);
    v16bf f;
#pragma unroll
    for (int i = 0; i < 8; ++i) { f[i] = lo[i]; f[i + 8] = hi[i]; }
    return f;
}

__device__ __forceinline__ float gelu_exact(float x) {
    return 0.5f * x * (1.0f + erff(x * 0.70710678118654752f));
}

// C/D layout: col = n + (lane&15); VGPR r -> row m + r + ((lane>>4)*8)
__device__ __forceinline__ void store_tile(v8f acc, int m, int n, int lane,
                                           const float* __restrict__ bias,
                                           float* __restrict__ outF,
                                           bf16_t* __restrict__ outB,
                                           int Nvalid, int act) {
    const int col = n + (lane & 15);
    if (col >= Nvalid) return;
    const int rbase = m + ((lane >> 4) << 3);
    const float bv = bias ? bias[col] : 0.0f;
#pragma unroll
    for (int r = 0; r < 8; ++r) {
        float v = acc[r] + bv;
        if (act == 1) v = gelu_exact(v);
        const size_t idx = (size_t)(rbase + r) * (size_t)Nvalid + col;
        if (outF) outF[idx] = v;
        if (outB) outB[idx] = (bf16_t)v;
    }
}

// ---------------------------------------------------------------------------
// GEMM: C[M,Nout] = A[M,K](bf16) * WT[Nout,K](bf16) + bias (opt GELU).
// Block 256 thr = 8 waves as 4(M)x2(N); wave tile 32x64 (2x4 WMMA tiles).
// Block tile 128x128, K slabs of 32 double-buffered through LDS (async fill).
// Grid: (Nout/128, M/128). Output ld = Nvalid; pad cols skipped.
// ---------------------------------------------------------------------------
__global__ void k_gemm_wmma(const bf16_t* __restrict__ A,
                            const bf16_t* __restrict__ WT,
                            const float*  __restrict__ bias,
                            float*  __restrict__ outF,
                            bf16_t* __restrict__ outB,
                            int M, int Nout, int K, int Nvalid,
                            int act, int causal) {
    __shared__ bf16_t sA[2][BT * TPAD];
    __shared__ bf16_t sB[2][BT * TPAD];

    const int tid  = threadIdx.x;
    const int lane = tid & 31;
    const int wid  = tid >> 5;
    const int wm   = (wid >> 1) * 32;   // 0,32,64,96   (M offset in block)
    const int wn   = (wid & 1) * 64;    // 0,64         (N offset in block)
    const int mblk = blockIdx.y * BT;
    const int nblk = blockIdx.x * BT;

    if (causal && nblk > mblk + (BT - 1)) return;  // fully-masked scores tile

    v8f acc[2][4] = {};

    const int nsteps = K / KT;
    fill_tile(sA[0], A,  mblk, K, 0, tid);
    fill_tile(sB[0], WT, nblk, K, 0, tid);

    for (int s = 0; s < nsteps; ++s) {
        const int cur = s & 1;
        wait_async_fills();
        __syncthreads();   // tiles ready; previous compute finished everywhere
        if (s + 1 < nsteps) {
            fill_tile(sA[cur ^ 1], A,  mblk, K, (s + 1) * KT, tid);
            fill_tile(sB[cur ^ 1], WT, nblk, K, (s + 1) * KT, tid);
        }
        v16bf a0 = lds_frag(sA[cur], wm,      lane);
        v16bf a1 = lds_frag(sA[cur], wm + 16, lane);
        v16bf b0 = lds_frag(sB[cur], wn,      lane);
        v16bf b1 = lds_frag(sB[cur], wn + 16, lane);
        v16bf b2 = lds_frag(sB[cur], wn + 32, lane);
        v16bf b3 = lds_frag(sB[cur], wn + 48, lane);
        acc[0][0] = __builtin_amdgcn_wmma_f32_16x16x32_bf16(false, a0, false, b0, (short)0, acc[0][0], false, false);
        acc[0][1] = __builtin_amdgcn_wmma_f32_16x16x32_bf16(false, a0, false, b1, (short)0, acc[0][1], false, false);
        acc[0][2] = __builtin_amdgcn_wmma_f32_16x16x32_bf16(false, a0, false, b2, (short)0, acc[0][2], false, false);
        acc[0][3] = __builtin_amdgcn_wmma_f32_16x16x32_bf16(false, a0, false, b3, (short)0, acc[0][3], false, false);
        acc[1][0] = __builtin_amdgcn_wmma_f32_16x16x32_bf16(false, a1, false, b0, (short)0, acc[1][0], false, false);
        acc[1][1] = __builtin_amdgcn_wmma_f32_16x16x32_bf16(false, a1, false, b1, (short)0, acc[1][1], false, false);
        acc[1][2] = __builtin_amdgcn_wmma_f32_16x16x32_bf16(false, a1, false, b2, (short)0, acc[1][2], false, false);
        acc[1][3] = __builtin_amdgcn_wmma_f32_16x16x32_bf16(false, a1, false, b3, (short)0, acc[1][3], false, false);
    }

#pragma unroll
    for (int i = 0; i < 2; ++i)
#pragma unroll
        for (int j = 0; j < 4; ++j)
            store_tile(acc[i][j], mblk + wm + i * 16, nblk + wn + j * 16,
                       lane, bias, outF, outB, Nvalid, act);
}

// ---------------------------------------------------------------------------
// Embedding: x[n,d] = wte[tok[n],d] + pe[n%S,d]; f32 + bf16 copies.
// ---------------------------------------------------------------------------
__global__ void k_embed(const int* __restrict__ tokens,
                        const float* __restrict__ wte,
                        const float* __restrict__ pe,
                        float* __restrict__ xf, bf16_t* __restrict__ xb) {
    const int n = blockIdx.x;
    const int s = n & (GPT_S - 1);
    const int t = tokens[n];
    for (int d = threadIdx.x; d < GPT_D; d += blockDim.x) {
        float v = wte[(size_t)t * GPT_D + d] + pe[(size_t)s * GPT_D + d];
        xf[(size_t)n * GPT_D + d] = v;
        xb[(size_t)n * GPT_D + d] = (bf16_t)v;
    }
}

// Weight convert+transpose: WT[n,k] = (bf16) W[k,n], zero-pad n >= Nsrc.
__global__ void k_cvt_wT(const float* __restrict__ W, bf16_t* __restrict__ WT,
                         int K, int Nsrc, int Npad) {
    const size_t i = (size_t)blockIdx.x * blockDim.x + threadIdx.x;
    const size_t total = (size_t)Npad * K;
    if (i >= total) return;
    const int n = (int)(i / K), k = (int)(i % K);
    WT[i] = (n < Nsrc) ? (bf16_t)W[(size_t)k * Nsrc + n] : (bf16_t)0.0f;
}

// vT[b,d,s] = v[b,s,d] (bf16), so V can be the WT operand of attn @ V.
__global__ void k_transpose_v(const bf16_t* __restrict__ v, bf16_t* __restrict__ vT) {
    const size_t i = (size_t)blockIdx.x * blockDim.x + threadIdx.x;
    const size_t total = (size_t)GPT_B * GPT_D * GPT_S;
    if (i >= total) return;
    const int b = (int)(i / ((size_t)GPT_D * GPT_S));
    const int rem = (int)(i % ((size_t)GPT_D * GPT_S));
    const int d = rem / GPT_S, s = rem % GPT_S;
    vT[i] = v[((size_t)b * GPT_S + s) * GPT_D + d];
}

// ---------------------------------------------------------------------------
// Causal softmax: attn[k] = softmax(scores[k<=q]/sqrt(D)); masked k never read.
// ---------------------------------------------------------------------------
__global__ void k_softmax_causal(const float* __restrict__ scores,
                                 bf16_t* __restrict__ attn) {
    __shared__ float red[256];
    const int row = blockIdx.x;            // b*S + q
    const int q = row & (GPT_S - 1);
    const float* srow = scores + (size_t)row * GPT_S;
    bf16_t* arow = attn + (size_t)row * GPT_S;
    const float scale = 0.03608439182435161f;  // 1/sqrt(768)
    const int tid = threadIdx.x;

    float mx = -INFINITY;
    for (int k = tid; k <= q; k += 256) mx = fmaxf(mx, srow[k] * scale);
    red[tid] = mx; __syncthreads();
    for (int s = 128; s > 0; s >>= 1) { if (tid < s) red[tid] = fmaxf(red[tid], red[tid + s]); __syncthreads(); }
    mx = red[0]; __syncthreads();

    float sum = 0.0f;
    for (int k = tid; k <= q; k += 256) sum += __expf(srow[k] * scale - mx);
    red[tid] = sum; __syncthreads();
    for (int s = 128; s > 0; s >>= 1) { if (tid < s) red[tid] += red[tid + s]; __syncthreads(); }
    const float inv = 1.0f / red[0];

    for (int k = tid; k < GPT_S; k += 256)
        arow[k] = (k <= q) ? (bf16_t)(__expf(srow[k] * scale - mx) * inv) : (bf16_t)0.0f;
}

// ---------------------------------------------------------------------------
// LayerNorm(in + res) * g + b ; optional f32/bf16 outputs. One block per row.
// ---------------------------------------------------------------------------
__global__ void k_layernorm(const float* __restrict__ in,
                            const float* __restrict__ res,
                            const float* __restrict__ g,
                            const float* __restrict__ bb,
                            float* __restrict__ outF, bf16_t* __restrict__ outB) {
    __shared__ float r1[256], r2[256];
    const int row = blockIdx.x, tid = threadIdx.x;
    const float* ir = in + (size_t)row * GPT_D;
    const float* rr = res ? res + (size_t)row * GPT_D : nullptr;

    float s = 0.0f, s2 = 0.0f;
    for (int d = tid; d < GPT_D; d += 256) {
        float v = ir[d] + (rr ? rr[d] : 0.0f);
        s += v; s2 += v * v;
    }
    r1[tid] = s; r2[tid] = s2; __syncthreads();
    for (int k = 128; k > 0; k >>= 1) {
        if (tid < k) { r1[tid] += r1[tid + k]; r2[tid] += r2[tid + k]; }
        __syncthreads();
    }
    const float mu = r1[0] * (1.0f / GPT_D);
    const float var = r2[0] * (1.0f / GPT_D) - mu * mu;
    const float rstd = rsqrtf(var + LN_EPS);

    for (int d = tid; d < GPT_D; d += 256) {
        float v = ir[d] + (rr ? rr[d] : 0.0f);
        float o = (v - mu) * rstd * g[d] + bb[d];
        if (outF) outF[(size_t)row * GPT_D + d] = o;
        if (outB) outB[(size_t)row * GPT_D + d] = (bf16_t)o;
    }
}

// ---------------------------------------------------------------------------
// Deterministic NLL: per-row -(logp[target])/N into row_loss, then tree-sum.
// ---------------------------------------------------------------------------
__global__ void k_nll_rows(const float* __restrict__ logits,
                           const int* __restrict__ targets,
                           float* __restrict__ row_loss) {
    __shared__ float red[256];
    const int row = blockIdx.x, tid = threadIdx.x;
    const float* lr = logits + (size_t)row * GPT_V;

    float mx = -INFINITY;
    for (int i = tid; i < GPT_V; i += 256) mx = fmaxf(mx, lr[i]);
    red[tid] = mx; __syncthreads();
    for (int s = 128; s > 0; s >>= 1) { if (tid < s) red[tid] = fmaxf(red[tid], red[tid + s]); __syncthreads(); }
    mx = red[0]; __syncthreads();

    float sum = 0.0f;
    for (int i = tid; i < GPT_V; i += 256) sum += __expf(lr[i] - mx);
    red[tid] = sum; __syncthreads();
    for (int s = 128; s > 0; s >>= 1) { if (tid < s) red[tid] += red[tid + s]; __syncthreads(); }

    if (tid == 0) {
        const float lse = mx + __logf(red[0]);
        row_loss[row] = -(lr[targets[row]] - lse) * (1.0f / GPT_N);
    }
}

__global__ void k_loss_reduce(const float* __restrict__ row_loss, float* __restrict__ loss) {
    __shared__ float red[256];
    const int tid = threadIdx.x;
    float s = 0.0f;
    for (int i = tid; i < GPT_N; i += 256) s += row_loss[i];
    red[tid] = s; __syncthreads();
    for (int k = 128; k > 0; k >>= 1) { if (tid < k) red[tid] += red[tid + k]; __syncthreads(); }
    if (tid == 0) *loss = red[0];
}

// ---------------------------------------------------------------------------
// Host launcher
// ---------------------------------------------------------------------------
extern "C" void kernel_launch(void* const* d_in, const int* in_sizes, int n_in,
                              void* d_out, int out_size, void* d_ws, size_t ws_size,
                              hipStream_t stream) {
    // setup_inputs() order: tokens, targets, params{wte,wq,bq,wk,bk,wv,bv,wo,bo,
    //                        w1,b1,w2,b2,ln1_g,ln1_b,ln2_g,ln2_b,w_lm,b_lm}, pe
    const int*   tokens  = (const int*)d_in[0];
    const int*   targets = (const int*)d_in[1];
    const float* wte  = (const float*)d_in[2];
    const float* wq   = (const float*)d_in[3];  const float* bq = (const float*)d_in[4];
    const float* wk   = (const float*)d_in[5];  const float* bk = (const float*)d_in[6];
    const float* wv   = (const float*)d_in[7];  const float* bv = (const float*)d_in[8];
    const float* wo   = (const float*)d_in[9];  const float* bo = (const float*)d_in[10];
    const float* w1   = (const float*)d_in[11]; const float* b1 = (const float*)d_in[12];
    const float* w2   = (const float*)d_in[13]; const float* b2 = (const float*)d_in[14];
    const float* ln1g = (const float*)d_in[15]; const float* ln1b = (const float*)d_in[16];
    const float* ln2g = (const float*)d_in[17]; const float* ln2b = (const float*)d_in[18];
    const float* wlm  = (const float*)d_in[19]; const float* blm = (const float*)d_in[20];
    const float* pe   = (const float*)d_in[21];

    float* logits = (float*)d_out;                       // [N, V] f32
    float* d_loss = logits + (size_t)GPT_N * GPT_V;      // scalar

    // ---- workspace bump allocator (256B aligned) ----
    char* wp = (char*)d_ws;
    auto alloc = [&](size_t bytes) -> void* {
        void* p = (void*)wp;
        wp += (bytes + 255) & ~(size_t)255;
        return p;
    };
    const size_t ND = (size_t)GPT_N * GPT_D, NH = (size_t)GPT_N * GPT_H;
    const size_t SS = (size_t)GPT_B * GPT_S * GPT_S, DS = (size_t)GPT_B * GPT_D * GPT_S;

    float*  x_f    = (float*) alloc(ND * 4);
    float*  ao_f   = (float*) alloc(ND * 4);
    float*  adn_f  = (float*) alloc(ND * 4);
    float*  h2_f   = (float*) alloc(ND * 4);
    float*  sc_f   = (float*) alloc(SS * 4);
    float*  rl_f   = (float*) alloc((size_t)GPT_N * 4);
    bf16_t* x_b    = (bf16_t*)alloc(ND * 2);
    bf16_t* q_b    = (bf16_t*)alloc(ND * 2);
    bf16_t* k_b    = (bf16_t*)alloc(ND * 2);
    bf16_t* v_b    = (bf16_t*)alloc(ND * 2);
    bf16_t* ctx_b  = (bf16_t*)alloc(ND * 2);
    bf16_t* adn_b  = (bf16_t*)alloc(ND * 2);
    bf16_t* h1_b   = (bf16_t*)alloc(NH * 2);
    bf16_t* h_b    = (bf16_t*)alloc(ND * 2);
    bf16_t* attn_b = (bf16_t*)alloc(SS * 2);
    bf16_t* vT_b   = (bf16_t*)alloc(DS * 2);
    bf16_t* wqT = (bf16_t*)alloc((size_t)GPT_D * GPT_D * 2);
    bf16_t* wkT = (bf16_t*)alloc((size_t)GPT_D * GPT_D * 2);
    bf16_t* wvT = (bf16_t*)alloc((size_t)GPT_D * GPT_D * 2);
    bf16_t* woT = (bf16_t*)alloc((size_t)GPT_D * GPT_D * 2);
    bf16_t* w1T = (bf16_t*)alloc((size_t)GPT_H * GPT_D * 2);
    bf16_t* w2T = (bf16_t*)alloc((size_t)GPT_D * GPT_H * 2);
    bf16_t* wlmT = (bf16_t*)alloc((size_t)GPT_VP * GPT_D * 2);
    (void)ws_size; (void)n_in; (void)in_sizes; (void)out_size;

    const int T = 256;
    auto cdiv = [](size_t a, size_t b) { return (unsigned)((a + b - 1) / b); };

    // ---- weight convert/transpose to bf16 ----
    k_cvt_wT<<<cdiv((size_t)GPT_D * GPT_D, T), T, 0, stream>>>(wq, wqT, GPT_D, GPT_D, GPT_D);
    k_cvt_wT<<<cdiv((size_t)GPT_D * GPT_D, T), T, 0, stream>>>(wk, wkT, GPT_D, GPT_D, GPT_D);
    k_cvt_wT<<<cdiv((size_t)GPT_D * GPT_D, T), T, 0, stream>>>(wv, wvT, GPT_D, GPT_D, GPT_D);
    k_cvt_wT<<<cdiv((size_t)GPT_D * GPT_D, T), T, 0, stream>>>(wo, woT, GPT_D, GPT_D, GPT_D);
    k_cvt_wT<<<cdiv((size_t)GPT_H * GPT_D, T), T, 0, stream>>>(w1, w1T, GPT_D, GPT_H, GPT_H);
    k_cvt_wT<<<cdiv((size_t)GPT_D * GPT_H, T), T, 0, stream>>>(w2, w2T, GPT_H, GPT_D, GPT_D);
    k_cvt_wT<<<cdiv((size_t)GPT_VP * GPT_D, T), T, 0, stream>>>(wlm, wlmT, GPT_D, GPT_V, GPT_VP);

    // ---- embedding ----
    k_embed<<<GPT_N, T, 0, stream>>>(tokens, wte, pe, x_f, x_b);

    // ---- QKV projections (bf16 out only) ----
    dim3 gqkv(GPT_D / BT, GPT_N / BT);
    k_gemm_wmma<<<gqkv, T, 0, stream>>>(x_b, wqT, bq, nullptr, q_b, GPT_N, GPT_D, GPT_D, GPT_D, 0, 0);
    k_gemm_wmma<<<gqkv, T, 0, stream>>>(x_b, wkT, bk, nullptr, k_b, GPT_N, GPT_D, GPT_D, GPT_D, 0, 0);
    k_gemm_wmma<<<gqkv, T, 0, stream>>>(x_b, wvT, bv, nullptr, v_b, GPT_N, GPT_D, GPT_D, GPT_D, 0, 0);

    // ---- scores = Q K^T (per batch; K rows already in WT layout) ----
    dim3 gsc(GPT_S / BT, GPT_S / BT);
    for (int b = 0; b < GPT_B; ++b)
        k_gemm_wmma<<<gsc, T, 0, stream>>>(q_b + (size_t)b * GPT_S * GPT_D,
                                           k_b + (size_t)b * GPT_S * GPT_D,
                                           nullptr, sc_f + (size_t)b * GPT_S * GPT_S, nullptr,
                                           GPT_S, GPT_S, GPT_D, GPT_S, 0, /*causal=*/1);

    // ---- causal softmax -> bf16 probabilities ----
    k_softmax_causal<<<GPT_B * GPT_S, T, 0, stream>>>(sc_f, attn_b);

    // ---- V transpose, ctx = attn @ V ----
    k_transpose_v<<<cdiv(DS, T), T, 0, stream>>>(v_b, vT_b);
    dim3 gctx(GPT_D / BT, GPT_S / BT);
    for (int b = 0; b < GPT_B; ++b)
        k_gemm_wmma<<<gctx, T, 0, stream>>>(attn_b + (size_t)b * GPT_S * GPT_S,
                                            vT_b + (size_t)b * GPT_D * GPT_S,
                                            nullptr, nullptr, ctx_b + (size_t)b * GPT_S * GPT_D,
                                            GPT_S, GPT_D, GPT_S, GPT_D, 0, 0);

    // ---- output projection (f32 for residual) ----
    k_gemm_wmma<<<gqkv, T, 0, stream>>>(ctx_b, woT, bo, ao_f, nullptr, GPT_N, GPT_D, GPT_D, GPT_D, 0, 0);

    // ---- LN1(x + att_out) ----
    k_layernorm<<<GPT_N, T, 0, stream>>>(x_f, ao_f, ln1g, ln1b, adn_f, adn_b);

    // ---- MLP ----
    dim3 gh1(GPT_H / BT, GPT_N / BT);
    k_gemm_wmma<<<gh1, T, 0, stream>>>(adn_b, w1T, b1, nullptr, h1_b, GPT_N, GPT_H, GPT_D, GPT_H, /*gelu*/1, 0);
    k_gemm_wmma<<<gqkv, T, 0, stream>>>(h1_b, w2T, b2, h2_f, nullptr, GPT_N, GPT_D, GPT_H, GPT_D, 0, 0);

    // ---- LN2(h + adn) -> bf16 for LM head ----
    k_layernorm<<<GPT_N, T, 0, stream>>>(h2_f, adn_f, ln2g, ln2b, nullptr, h_b);

    // ---- LM head: logits straight into d_out (ld = V, pad cols skipped) ----
    dim3 glm(GPT_VP / BT, GPT_N / BT);
    k_gemm_wmma<<<glm, T, 0, stream>>>(h_b, wlmT, blm, logits, nullptr,
                                       GPT_N, GPT_VP, GPT_D, GPT_V, 0, 0);

    // ---- deterministic loss ----
    k_nll_rows<<<GPT_N, T, 0, stream>>>(logits, targets, rl_f);
    k_loss_reduce<<<1, T, 0, stream>>>(rl_f, d_loss);
}
